// RGCN_90795608637584
// MI455X (gfx1250) — matrix operand order repaired and checked by reading, not compiled, not run on hardware
//
#include <hip/hip_runtime.h>

// ---------------------------------------------------------------------------
// RGCN (2 layers, mean aggregation, BN+ReLU between) for MI455X / gfx1250.
// Strategy: bf16 packed-atomic scatter aggregation (L2-resident agg buffer),
// grouped GEMM via v_wmma_f32_16x16x32_bf16, f32 accumulate.
// ---------------------------------------------------------------------------

#define NREL 8
#define BN_EPS 1e-5f

typedef __attribute__((ext_vector_type(16))) __bf16 v16bf;
typedef __attribute__((ext_vector_type(8)))  float  v8f;

union ABFrag {
    uint4 q[2];
    v16bf v;
};

__device__ __forceinline__ unsigned pack_bf16(float a, float b) {
    unsigned ua = __builtin_bit_cast(unsigned, a);
    unsigned ub = __builtin_bit_cast(unsigned, b);
    ua += 0x7fffu + ((ua >> 16) & 1u);   // round to nearest even
    ub += 0x7fffu + ((ub >> 16) & 1u);
    return (ua >> 16) | (ub & 0xffff0000u);
}
__device__ __forceinline__ float bf_lo(unsigned d) {
    return __builtin_bit_cast(float, d << 16);
}
__device__ __forceinline__ float bf_hi(unsigned d) {
    return __builtin_bit_cast(float, d & 0xffff0000u);
}
__device__ __forceinline__ unsigned short f2bf(float a) {
    unsigned ua = __builtin_bit_cast(unsigned, a);
    ua += 0x7fffu + ((ua >> 16) & 1u);
    return (unsigned short)(ua >> 16);
}

// ---------------------------------------------------------------------------
// Scatter: per edge, pk-bf16 atomic-add of the 64-ch source row into
// agg[dst*NREL + etype], plus a degree counter. 8 threads/edge, 8 ch each.
// ---------------------------------------------------------------------------
__global__ __launch_bounds__(256)
void rgcn_scatter(const float* __restrict__ xf,            // f32 source (layer1) or null
                  const unsigned* __restrict__ xb,         // bf16-pair source (layer2) or null
                  const int* __restrict__ ei,              // [2*E] src then dst
                  const int* __restrict__ et,              // [E]
                  unsigned* __restrict__ agg,              // bf16 pairs [N*NREL*32]
                  float* __restrict__ cnt,                 // [N*NREL]
                  int E)
{
    long gid = (long)blockIdx.x * 256 + threadIdx.x;
    int e = (int)(gid >> 3);
    int part = (int)(gid & 7);
    if (e >= E) return;

    int src = ei[e];
    int dst = ei[(size_t)E + e];
    int t   = et[e];

    unsigned pk[4];
    if (xb) {
        const uint4* hq = (const uint4*)xb;
        uint4 d = hq[(size_t)src * 8 + part];
        pk[0] = d.x; pk[1] = d.y; pk[2] = d.z; pk[3] = d.w;
    } else {
        const float4* x4 = (const float4*)xf;
        float4 a = x4[(size_t)src * 16 + part * 2];
        float4 b = x4[(size_t)src * 16 + part * 2 + 1];
        pk[0] = pack_bf16(a.x, a.y);
        pk[1] = pack_bf16(a.z, a.w);
        pk[2] = pack_bf16(b.x, b.y);
        pk[3] = pack_bf16(b.z, b.w);
    }

    unsigned* d0 = agg + ((size_t)dst * NREL + t) * 32 + part * 4;
#pragma unroll
    for (int i = 0; i < 4; ++i) {
        asm volatile("global_atomic_pk_add_bf16 %0, %1, off"
                     :: "v"(d0 + i), "v"(pk[i]) : "memory");
    }
    if (part == 0) {
        atomicAdd(&cnt[(size_t)dst * NREL + t], 1.0f);
    }
}

// ---------------------------------------------------------------------------
// Grouped GEMM: out[16x64 tile] = sum_r mean_r(16x64) @ W[r] + root_in @ Wroot
// One block = 64 nodes = 4 waves; each wave owns a 16-node tile.
// A and B staged in LDS in WMMA fragment order (stride-36 dword padding).
// ---------------------------------------------------------------------------
__global__ __launch_bounds__(128)
void rgcn_gemm(const unsigned* __restrict__ agg,    // bf16 pairs [N*NREL*32]
               const float* __restrict__ cnt,       // [N*NREL]
               const float* __restrict__ rootf,     // f32 root input (layer1) or null
               const unsigned* __restrict__ rootb,  // bf16-pair root input (layer2) or null
               const float* __restrict__ W,         // [NREL*64*64] f32
               const float* __restrict__ Wroot,     // [64*64] f32
               const float* __restrict__ bias,      // [64] f32
               unsigned short* __restrict__ outb,   // bf16 out (layer1) or null
               float* __restrict__ outf,            // f32 out (layer2) or null
               int N)
{
    __shared__ float    s_inv[64];
    __shared__ unsigned sA[64 * 36];
    __shared__ unsigned sB[64 * 36];

    const int tid  = threadIdx.x;
    const int lane = tid & 31;
    const int w    = tid >> 5;
    const int tileBase = blockIdx.x * 64;

    v8f acc[4] = {};

    for (int rel = 0; rel <= NREL; ++rel) {
        __syncthreads();                       // previous iter's compute done
        if (rel < NREL && tid < 64) {
            int node = tileBase + tid;
            float c = (node < N) ? cnt[(size_t)node * NREL + rel] : 1.0f;
            // fast reciprocal: divisor is a small positive integer count
            s_inv[tid] = __builtin_amdgcn_rcpf(fmaxf(c, 1.0f));
        }
        __syncthreads();

        // ---- stage A (64 nodes x 64 K, bf16) and B (W[rel], 64x64 bf16) ----
        const float* Wp = (rel < NREL) ? (W + (size_t)rel * 4096) : Wroot;
#pragma unroll
        for (int it = 0; it < 16; ++it) {
            int q    = it * 128 + tid;         // 0..2047 dwords
            int node = q >> 5;                 // 0..63
            int kk2  = q & 31;                 // channel-pair index (K/2)
            // fragment position: contiguous K (dword kk2 -> k 2*kk2, 2*kk2+1)
            int posA = kk2;
            unsigned dA = 0;
            int gnode = tileBase + node;
            if (gnode < N) {
                if (rel < NREL) {
                    unsigned d = agg[((size_t)gnode * NREL + rel) * 32 + kk2];
                    float inv = s_inv[node];
                    dA = pack_bf16(bf_lo(d) * inv, bf_hi(d) * inv);
                } else if (rootf) {
                    float2 xv = ((const float2*)rootf)[(size_t)gnode * 32 + kk2];
                    dA = pack_bf16(xv.x, xv.y);
                } else {
                    dA = rootb[(size_t)gnode * 32 + kk2];
                }
            }
            sA[node * 36 + posA] = dA;

            // B: dword (col c, step, halfB, v) = pack(W[k0][c], W[k0+1][c])
            int c     = q >> 5;                // col 0..63
            int r32   = q & 31;
            int step  = r32 >> 4;
            int halfB = (r32 >> 3) & 1;
            int v     = r32 & 7;
            int k0    = step * 32 + halfB * 16 + 2 * v;
            unsigned dB = pack_bf16(Wp[k0 * 64 + c], Wp[(k0 + 1) * 64 + c]);
            sB[c * 36 + step * 16 + halfB * 8 + v] = dB;
        }
        __syncthreads();

        // ---- compute: 2 K-steps x 4 N-tiles of v_wmma_f32_16x16x32_bf16 ----
        const int m    = lane & 15;
        const int half = lane >> 4;
        const int arow = w * 16 + m;
#pragma unroll
        for (int s = 0; s < 2; ++s) {
            ABFrag a;
            const uint4* pa = (const uint4*)&sA[arow * 36 + s * 16 + half * 8];
            a.q[0] = pa[0];
            a.q[1] = pa[1];
#pragma unroll
            for (int nt = 0; nt < 4; ++nt) {
                ABFrag b;
                const uint4* pb = (const uint4*)&sB[(nt * 16 + m) * 36 + s * 16 + half * 8];
                b.q[0] = pb[0];
                b.q[1] = pb[1];
                acc[nt] = __builtin_amdgcn_wmma_f32_16x16x32_bf16(
                    false, a.v, false, b.v, (short)0, acc[nt], false, false);
            }
        }
    }

    // ---- epilogue: C layout = lane(0-15|16-31) cols, VGPR j -> row j | j+8 ----
    const int m    = lane & 15;
    const int half = lane >> 4;
#pragma unroll
    for (int nt = 0; nt < 4; ++nt) {
        int col = nt * 16 + m;
        float bcol = bias[col];
#pragma unroll
        for (int j = 0; j < 8; ++j) {
            int node = tileBase + w * 16 + half * 8 + j;
            if (node < N) {
                float val = acc[nt][j] + bcol;
                if (outf) outf[(size_t)node * 64 + col] = val;
                else      outb[(size_t)node * 64 + col] = f2bf(val);
            }
        }
    }
}

// ---------------------------------------------------------------------------
// BatchNorm stats: per-channel sum & sumsq over N rows (two-stage reduce).
// ---------------------------------------------------------------------------
__global__ __launch_bounds__(256)
void bn_reduce(const unsigned short* __restrict__ h, float* __restrict__ stats, int N)
{
    __shared__ float ss[256];
    __shared__ float ss2[256];
    int tid = threadIdx.x;
    int c = tid & 63, g = tid >> 6;
    float s = 0.f, s2 = 0.f;
    for (int row = blockIdx.x * 4 + g; row < N; row += 256 * 4) {
        float v = __builtin_bit_cast(float, (unsigned)h[(size_t)row * 64 + c] << 16);
        s += v;
        s2 += v * v;
    }
    ss[tid] = s;
    ss2[tid] = s2;
    __syncthreads();
    if (g == 0) {
        s  = ss[c]  + ss[64 + c]  + ss[128 + c]  + ss[192 + c];
        s2 = ss2[c] + ss2[64 + c] + ss2[128 + c] + ss2[192 + c];
        atomicAdd(&stats[c], s);
        atomicAdd(&stats[64 + c], s2);
    }
}

// ---------------------------------------------------------------------------
// BN apply + ReLU, in-place on bf16 h (dword = 2 channels).
// ---------------------------------------------------------------------------
__global__ __launch_bounds__(256)
void bn_apply(unsigned* __restrict__ hdw, const float* __restrict__ stats,
              const float* __restrict__ gamma, const float* __restrict__ beta, int N)
{
    size_t gid = (size_t)blockIdx.x * 256 + threadIdx.x;
    if (gid >= (size_t)N * 32) return;
    int c0 = (int)(gid & 31) * 2;
    unsigned d = hdw[gid];
    float invN = 1.0f / (float)N;
    float out01[2];
    float vv[2] = {bf_lo(d), bf_hi(d)};
#pragma unroll
    for (int i = 0; i < 2; ++i) {
        int c = c0 + i;
        float mu  = stats[c] * invN;
        float var = stats[64 + c] * invN - mu * mu;
        float scl = gamma[c] * rsqrtf(var + BN_EPS);
        float v   = (vv[i] - mu) * scl + beta[c];
        out01[i]  = fmaxf(v, 0.0f);
    }
    hdw[gid] = pack_bf16(out01[0], out01[1]);
}

// ---------------------------------------------------------------------------
extern "C" void kernel_launch(void* const* d_in, const int* in_sizes, int n_in,
                              void* d_out, int out_size, void* d_ws, size_t ws_size,
                              hipStream_t stream)
{
    (void)n_in; (void)out_size; (void)ws_size;
    const float* x      = (const float*)d_in[0];
    const int*   ei     = (const int*)d_in[1];
    const int*   et     = (const int*)d_in[2];
    const float* w1     = (const float*)d_in[3];
    const float* root1  = (const float*)d_in[4];
    const float* b1     = (const float*)d_in[5];
    const float* gamma1 = (const float*)d_in[6];
    const float* beta1  = (const float*)d_in[7];
    const float* w2     = (const float*)d_in[8];
    const float* root2  = (const float*)d_in[9];
    const float* b2     = (const float*)d_in[10];
    float* out = (float*)d_out;

    const int N = in_sizes[0] / 64;
    const int E = in_sizes[2];

    char* ws = (char*)d_ws;
    size_t aggBytes = (size_t)N * NREL * 64 * 2;   // bf16 agg
    size_t cntBytes = (size_t)N * NREL * 4;        // f32 degree counts
    size_t hBytes   = (size_t)N * 64 * 2;          // bf16 hidden
    unsigned*       agg   = (unsigned*)ws;
    float*          cnt   = (float*)(ws + aggBytes);
    unsigned short* h     = (unsigned short*)(ws + aggBytes + cntBytes);
    float*          stats = (float*)(ws + aggBytes + cntBytes + hBytes);

    const int sblocks = (int)(((long)E * 8 + 255) / 256);
    const int gblocks = (N + 63) / 64;
    const int ablocks = (int)(((size_t)N * 32 + 255) / 256);

    // -------- layer 1 --------
    hipMemsetAsync(agg, 0, aggBytes, stream);
    hipMemsetAsync(cnt, 0, cntBytes, stream);
    hipMemsetAsync(stats, 0, 128 * sizeof(float), stream);
    rgcn_scatter<<<sblocks, 256, 0, stream>>>(x, nullptr, ei, et, agg, cnt, E);
    rgcn_gemm<<<gblocks, 128, 0, stream>>>(agg, cnt, x, nullptr,
                                           w1, root1, b1, h, nullptr, N);
    bn_reduce<<<256, 256, 0, stream>>>(h, stats, N);
    bn_apply<<<ablocks, 256, 0, stream>>>((unsigned*)h, stats, gamma1, beta1, N);

    // -------- layer 2 --------
    hipMemsetAsync(agg, 0, aggBytes, stream);
    hipMemsetAsync(cnt, 0, cntBytes, stream);
    rgcn_scatter<<<sblocks, 256, 0, stream>>>(nullptr, (const unsigned*)h, ei, et, agg, cnt, E);
    rgcn_gemm<<<gblocks, 128, 0, stream>>>(agg, cnt, nullptr, (const unsigned*)h,
                                           w2, root2, b2, nullptr, out, N);
}
